// QALanguageModel_82806969467029
// MI455X (gfx1250) — matrix operand order repaired and checked by reading, not compiled, not run on hardware
//
#include <hip/hip_runtime.h>
#include <math.h>

// ---------------- Types for CDNA5 WMMA ----------------
typedef _Float16 half_t;
typedef __attribute__((ext_vector_type(16))) _Float16 v16h;
typedef __attribute__((ext_vector_type(8)))  float    v8f;
typedef __attribute__((ext_vector_type(4)))  float    f4v;   // 16-byte vector for LDS/global moves

union Frag16 { v16h v; f4v f4[2]; };

__device__ __forceinline__ v8f wmma_f16(v16h a, v16h b, v8f c) {
  // D = A(16x32) * B(32x16) + C(16x16 f32); args: neg_a,A,neg_b,B,c_mod,C,reuse_a,reuse_b
  return __builtin_amdgcn_wmma_f32_16x16x32_f16(false, a, false, b, (short)0, c, false, false);
}

__device__ __forceinline__ float gelu_exact(float x) {
  return 0.5f * x * (1.0f + erff(x * 0.70710678118654752f));
}

// ---- CDNA5 async global->LDS staging (ASYNCcnt path), with sync fallback ----
#if __has_builtin(__builtin_amdgcn_global_load_async_to_lds_b128)
#define USE_ASYNC_LDS 1
#else
#define USE_ASYNC_LDS 0
#endif

// Builtin signature (from clang diagnostic): arg0 = GCC-vector int4 pointer in
// addrspace(1) (printed "__device__"), arg1 = LDS-side pointer, then two imm ints.
typedef int v4i_t __attribute__((vector_size(16)));
typedef __attribute__((address_space(1))) v4i_t* gvec_ptr;
typedef __attribute__((address_space(3))) v4i_t* lvec_ptr;
#define GAS1(p) ((gvec_ptr)(v4i_t*)(void*)(p))
#define LAS3(p) ((lvec_ptr)(v4i_t*)(void*)(p))

__device__ __forceinline__ void wait_async_all() {
#if USE_ASYNC_LDS
#if __has_builtin(__builtin_amdgcn_s_wait_asynccnt)
  __builtin_amdgcn_s_wait_asynccnt(0);
#else
  asm volatile("s_wait_asynccnt 0x0" ::: "memory");
#endif
#endif
}

// ---------------- Model dims ----------------
#define HH     768
#define LL     12
#define NHEAD  12
#define DHEAD  64
#define FFD    3072
#define VOCAB  30522
#define BB     8
#define SS     512
#define MM     (BB*SS)   // 4096 tokens

// ---------------- Generic batched WMMA GEMM ----------------
// C[z] = act( alpha * A[z](MxK,f16,row-major,lda) * B[z]^T((NxK),f16,row-major,ldb) + bias + resid )
// Per-batch bases: z -> (bo=z/binner, bi=z%binner); base = bo*O + bi*I (elements).
#define BM 64
#define BN 128
#define BK 32
#define LDSS 48   // LDS row stride in halfs (96B: 16B-aligned, staggers banks)

__global__ __launch_bounds__(256) void gemm_wmma_kernel(
    const half_t* __restrict__ A, int lda,
    const half_t* __restrict__ BT, int ldb,
    float* __restrict__ Cf32, half_t* __restrict__ Cf16, int ldc,
    const float* __restrict__ bias, const float* __restrict__ resid,
    int Mtot, int Ntot, int K, float alpha, int epi /*0=none,1=gelu,2=resid-add*/,
    int causal, int vt_mode,
    int binner, long long aO, long long aI, long long bO, long long bI,
    long long cO, long long cI)
{
  __shared__ __align__(16) half_t As[2][BM * LDSS];
  __shared__ __align__(16) half_t Bs[2][BN * LDSS];
  const int tid  = threadIdx.x;
  const int lane = tid & 31;
  const int wave = tid >> 5;       // 0..7
  const int wm   = wave >> 2;      // 0..1  (32-row slab)
  const int wn   = wave & 3;       // 0..3  (32-col slab)
  const int z    = blockIdx.z;
  const int bo   = z / binner, bi = z % binner;
  const half_t* Ab = A  + (size_t)bo * aO + (size_t)bi * aI;
  const half_t* Bb = BT + (size_t)bo * bO + (size_t)bi * bI;
  const size_t cbase = (size_t)bo * cO + (size_t)bi * cI;
  const int m0 = blockIdx.y * BM;
  const int n0 = blockIdx.x * BN;
  const int half_ = lane >> 4;
  const int l16   = lane & 15;

  // --- staging geometry: per-thread 16B chunks; row indices clamped (never
  // branched): out-of-range A rows / B^T rows only feed C rows/cols that the
  // epilogue refuses to store, so clamped garbage is harmless.
  const int rowA = tid >> 2, segA = tid & 3;                 // 64 x 4 chunks (A: 4KB)
  int gmA = m0 + rowA; if (gmA > Mtot - 1) gmA = Mtot - 1;
  const half_t* gA = Ab + (size_t)gmA * lda + segA * 8;
  const int tB0 = tid, tB1 = tid + 256;                      // 512 chunks (B: 8KB)
  const int rowB0 = tB0 >> 2, segB0 = tB0 & 3;
  const int rowB1 = tB1 >> 2, segB1 = tB1 & 3;
  int gn0 = n0 + rowB0; if (gn0 > Ntot - 1) gn0 = Ntot - 1;
  int gn1 = n0 + rowB1; if (gn1 > Ntot - 1) gn1 = Ntot - 1;
  const half_t* gB0 = Bb + (size_t)gn0 * ldb + segB0 * 8;
  const half_t* gB1 = Bb + (size_t)gn1 * ldb + segB1 * 8;

  auto stage = [&](int k0, int buf) {
#if USE_ASYNC_LDS
    __builtin_amdgcn_global_load_async_to_lds_b128(GAS1(gA  + k0), LAS3(&As[buf][rowA  * LDSS + segA  * 8]), 0, 0);
    __builtin_amdgcn_global_load_async_to_lds_b128(GAS1(gB0 + k0), LAS3(&Bs[buf][rowB0 * LDSS + segB0 * 8]), 0, 0);
    __builtin_amdgcn_global_load_async_to_lds_b128(GAS1(gB1 + k0), LAS3(&Bs[buf][rowB1 * LDSS + segB1 * 8]), 0, 0);
#else
    f4v av = *(const f4v*)(gA + k0);
    f4v b0 = *(const f4v*)(gB0 + k0);
    f4v b1 = *(const f4v*)(gB1 + k0);
    *(f4v*)(&As[buf][rowA  * LDSS + segA  * 8]) = av;
    *(f4v*)(&Bs[buf][rowB0 * LDSS + segB0 * 8]) = b0;
    *(f4v*)(&Bs[buf][rowB1 * LDSS + segB1 * 8]) = b1;
#endif
  };

  const v8f vzero = {0.f,0.f,0.f,0.f,0.f,0.f,0.f,0.f};
  v8f acc[2][2];
  acc[0][0]=vzero; acc[0][1]=vzero; acc[1][0]=vzero; acc[1][1]=vzero;

  // prologue: fill buffer 0
  stage(0, 0);
  wait_async_all();
  __syncthreads();

  int cur = 0;
  for (int k0 = 0; k0 < K; k0 += BK) {
    if (k0 + BK < K) stage(k0 + BK, cur ^ 1);   // async fill of next buffer overlaps WMMA

    Frag16 af[2], bf[2];
#pragma unroll
    for (int i = 0; i < 2; ++i) {   // A frag: k = half*8 + 0..7 and 16+half*8+0..7
      const half_t* ar = &As[cur][(wm*32 + i*16 + l16) * LDSS];
      af[i].f4[0] = *(const f4v*)(ar + half_ * 8);
      af[i].f4[1] = *(const f4v*)(ar + 16 + half_ * 8);
    }
#pragma unroll
    for (int j = 0; j < 2; ++j) {   // B frag: k = half*8 + 0..7 and 16+half*8+0..7
      const half_t* br = &Bs[cur][(wn*32 + j*16 + l16) * LDSS];
      bf[j].f4[0] = *(const f4v*)(br + half_ * 8);
      bf[j].f4[1] = *(const f4v*)(br + 16 + half_ * 8);
    }
#pragma unroll
    for (int i = 0; i < 2; ++i)
#pragma unroll
      for (int j = 0; j < 2; ++j)
        acc[i][j] = wmma_f16(af[i].v, bf[j].v, acc[i][j]);

    wait_async_all();     // my fills of next buffer done
    __syncthreads();      // everyone's fills done + everyone's reads of cur drained
    cur ^= 1;
  }

  // epilogue; D layout: row m = v + 8*half, col n = lane&15
#pragma unroll
  for (int i = 0; i < 2; ++i) {
#pragma unroll
    for (int j = 0; j < 2; ++j) {
      int n = n0 + wn*32 + j*16 + l16;
      if (n >= Ntot) continue;
      float bvv = bias ? bias[n] : 0.f;
#pragma unroll
      for (int v = 0; v < 8; ++v) {
        int m = m0 + wm*32 + i*16 + v + 8*half_;
        if (m >= Mtot) continue;
        float val = acc[i][j][v] * alpha + bvv;
        if (causal && n > m) val -= 10000.f;     // (1-tril)*-10000
        if (epi == 1) val = gelu_exact(val);
        size_t offc = cbase + (size_t)m * ldc + n;
        if (epi == 2) val += resid[offc];
        if (vt_mode) { // write as V^T[b][head][d][s] for attn@V
          int b_ = m >> 9, s_ = m & 511, hd_ = n >> 6, d_ = n & 63;
          Cf16[(((size_t)(b_*NHEAD + hd_) * DHEAD + d_) << 9) + s_] = (half_t)val;
        } else {
          if (Cf32) Cf32[offc] = val;
          if (Cf16) Cf16[offc] = (half_t)val;
        }
      }
    }
  }
}

// ---------------- fp32 -> f16 transposed weight convert ----------------
// in[K][N] f32 -> out[N][K] f16; grid.z batches with element strides
__global__ __launch_bounds__(256) void transpose_f32_to_f16(
    const float* __restrict__ in, half_t* __restrict__ out,
    int K, int N, long long inStride, long long outStride)
{
  __shared__ float tile[32][33];
  const size_t zin  = (size_t)blockIdx.z * inStride;
  const size_t zout = (size_t)blockIdx.z * outStride;
  int n0 = blockIdx.x * 32, k0 = blockIdx.y * 32;
  int tx = threadIdx.x, ty = threadIdx.y;  // (32,8)
  for (int i = 0; i < 32; i += 8) {
    int k = k0 + ty + i, n = n0 + tx;
    tile[ty + i][tx] = (k < K && n < N) ? in[zin + (size_t)k * N + n] : 0.f;
  }
  __syncthreads();
  for (int i = 0; i < 32; i += 8) {
    int n = n0 + ty + i, k = k0 + tx;
    if (n < N && k < K) out[zout + (size_t)n * K + k] = (half_t)tile[tx][ty + i];
  }
}

// ---------------- Embedding gather (pre-LN) ----------------
__global__ __launch_bounds__(256) void embed_kernel(
    const int* __restrict__ xids, const float* __restrict__ we,
    const float* __restrict__ pe, const float* __restrict__ te,
    float* __restrict__ out)
{
  int m = blockIdx.x;                 // token 0..4095
  int tok = xids[m];
  int s = m & (SS - 1);
  const float* wr = we + (size_t)tok * HH;
  const float* pr = pe + (size_t)s * HH;
  float* o = out + (size_t)m * HH;
  for (int c = threadIdx.x; c < HH; c += 256) o[c] = wr[c] + pr[c] + te[c];
}

// ---------------- Row LayerNorm (one wave32 per row), dual f32+f16 output ----------------
__global__ __launch_bounds__(256) void layernorm_kernel(
    const float* __restrict__ x, const float* __restrict__ g, const float* __restrict__ b,
    float* __restrict__ y32, half_t* __restrict__ y16, int rows, int cols)
{
  int wave = threadIdx.x >> 5, lane = threadIdx.x & 31;
  int row = blockIdx.x * 8 + wave;
  if (row >= rows) return;
  const float* xr = x + (size_t)row * cols;
  float s = 0.f;
  for (int c = lane; c < cols; c += 32) s += xr[c];
  for (int msk = 16; msk; msk >>= 1) s += __shfl_xor(s, msk, 32);
  float mean = s / cols;
  float var = 0.f;
  for (int c = lane; c < cols; c += 32) { float d = xr[c] - mean; var += d * d; }
  for (int msk = 16; msk; msk >>= 1) var += __shfl_xor(var, msk, 32);
  float rstd = rsqrtf(var / cols + 1e-12f);
  float* yr = y32 + (size_t)row * cols;
  half_t* hr = y16 + (size_t)row * cols;
  for (int c = lane; c < cols; c += 32) {
    float o = (xr[c] - mean) * rstd * g[c] + b[c];
    yr[c] = o; hr[c] = (half_t)o;
  }
}

// ---------------- Row softmax f32->f16 (one wave32 per row) ----------------
__global__ __launch_bounds__(256) void softmax_kernel(
    const float* __restrict__ x, half_t* __restrict__ y, int rows, int cols)
{
  int wave = threadIdx.x >> 5, lane = threadIdx.x & 31;
  int row = blockIdx.x * 8 + wave;
  if (row >= rows) return;
  const float* xr = x + (size_t)row * cols;
  float mx = -1e30f;
  for (int c = lane; c < cols; c += 32) mx = fmaxf(mx, xr[c]);
  for (int msk = 16; msk; msk >>= 1) mx = fmaxf(mx, __shfl_xor(mx, msk, 32));
  float s = 0.f;
  for (int c = lane; c < cols; c += 32) s += __expf(xr[c] - mx);
  for (int msk = 16; msk; msk >>= 1) s += __shfl_xor(s, msk, 32);
  float inv = 1.0f / s;
  half_t* yr = y + (size_t)row * cols;
  for (int c = lane; c < cols; c += 32) yr[c] = (half_t)(__expf(xr[c] - mx) * inv);
}

// ---------------- Fused classifier GEMM + online log-sum-exp NLL ----------------
// 16 token rows per workgroup resident in LDS; stream W^T (V x 768) in 128-col chunks via WMMA.
#define AT_STRIDE 776   // 768 + 8 halfs pad; 16B-aligned row stride
__global__ __launch_bounds__(256) void cls_nll_kernel(
    const half_t* __restrict__ Hf, const half_t* __restrict__ WT,
    const float* __restrict__ clsb, const int* __restrict__ y,
    float* __restrict__ nll_out, int K, int Vn)
{
  __shared__ __align__(16) half_t Atile[16 * AT_STRIDE];
  __shared__ float chunk[16][129];
  __shared__ float part[16][8];
  __shared__ float mrun[16], srun[16], lyv[16], mnew_s[16];
  __shared__ int   ytg[16];

  int tid = threadIdx.x, lane = tid & 31, wave = tid >> 5;
  int row0 = blockIdx.x * 16;
  { // load 16 x 384 halfs (48 16B chunks per row) of hidden states
    int r = tid >> 4, sg = tid & 15;
    const f4v* src = (const f4v*)(Hf + (size_t)(row0 + r) * K);
    f4v* dst = (f4v*)&Atile[r * AT_STRIDE];
    for (int i = 0; i < 3; ++i) dst[sg + i * 16] = src[sg + i * 16];
  }
  if (tid < 16) { mrun[tid] = -1e30f; srun[tid] = 0.f; lyv[tid] = 0.f; ytg[tid] = y[row0 + tid]; }
  __syncthreads();

  const int half_ = lane >> 4, l16 = lane & 15;
  const v8f vzero = {0.f,0.f,0.f,0.f,0.f,0.f,0.f,0.f};
  int nchunks = (Vn + 127) / 128;
  for (int ch = 0; ch < nchunks; ++ch) {
    int n = ch * 128 + wave * 16 + l16;
    int nc = (n < Vn) ? n : (Vn - 1);
    const half_t* brow = WT + (size_t)nc * K;
    // prefetch next chunk's weight row (global_prefetch_b8)
    int n_next = n + 128;
    if (n_next < Vn) __builtin_prefetch(WT + (size_t)n_next * K, 0, 1);
    v8f acc = vzero;
    for (int k0 = 0; k0 < K; k0 += 32) {
      Frag16 af, bf;
      const half_t* ar = &Atile[l16 * AT_STRIDE + k0];
      af.f4[0] = *(const f4v*)(ar + half_ * 8);
      af.f4[1] = *(const f4v*)(ar + 16 + half_ * 8);
      bf.f4[0] = *(const f4v*)(brow + k0 + half_ * 8);
      bf.f4[1] = *(const f4v*)(brow + k0 + 16 + half_ * 8);
      acc = wmma_f16(af.v, bf.v, acc);
    }
    float bb = (n < Vn) ? clsb[n] : 0.f;
#pragma unroll
    for (int v = 0; v < 8; ++v)
      chunk[v + 8 * half_][wave * 16 + l16] = (n < Vn) ? (acc[v] + bb) : -1e30f;
    __syncthreads();

    if (tid < 128) {                      // per-row partial max over 16 cols
      int r = tid >> 3, sg = tid & 7;
      float mx = -1e30f;
      for (int i = 0; i < 16; ++i) mx = fmaxf(mx, chunk[r][sg * 16 + i]);
      part[r][sg] = mx;
    }
    __syncthreads();
    if (tid < 16) {
      float mx = mrun[tid];
      for (int i = 0; i < 8; ++i) mx = fmaxf(mx, part[tid][i]);
      mnew_s[tid] = mx;
      int t = ytg[tid] - ch * 128;
      if (t >= 0 && t < 128) lyv[tid] = chunk[tid][t];
    }
    __syncthreads();
    if (tid < 128) {                      // per-row partial sum-exp vs new max
      int r = tid >> 3, sg = tid & 7;
      float mn = mnew_s[r], s = 0.f;
      for (int i = 0; i < 16; ++i) s += __expf(chunk[r][sg * 16 + i] - mn);
      part[r][sg] = s;
    }
    __syncthreads();
    if (tid < 16) {
      float mn = mnew_s[tid], s = 0.f;
      for (int i = 0; i < 8; ++i) s += part[tid][i];
      srun[tid] = srun[tid] * __expf(mrun[tid] - mn) + s;
      mrun[tid] = mn;
    }
    __syncthreads();
  }
  if (tid < 16) nll_out[row0 + tid] = mrun[tid] + __logf(srun[tid]) - lyv[tid];
}

// ---------------- Deterministic mean reduction ----------------
__global__ __launch_bounds__(256) void reduce_mean_kernel(
    const float* __restrict__ nll, float* __restrict__ out, int n)
{
  __shared__ float s[256];
  float acc = 0.f;
  for (int i = threadIdx.x; i < n; i += 256) acc += nll[i];
  s[threadIdx.x] = acc; __syncthreads();
  for (int st = 128; st; st >>= 1) {
    if (threadIdx.x < st) s[threadIdx.x] += s[threadIdx.x + st];
    __syncthreads();
  }
  if (threadIdx.x == 0) out[0] = s[0] / (float)n;
}

// ---------------- Host side ----------------
static void launch_gemm(hipStream_t stream,
    const half_t* A, int lda, const half_t* BT, int ldb,
    float* Cf32, half_t* Cf16, int ldc,
    const float* bias, const float* resid,
    int M, int N, int K, float alpha, int epi, int causal, int vt,
    int batches, int binner,
    long long aO, long long aI, long long bO, long long bI, long long cO, long long cI)
{
  dim3 grid((N + BN - 1) / BN, (M + BM - 1) / BM, batches);
  gemm_wmma_kernel<<<grid, dim3(256), 0, stream>>>(A, lda, BT, ldb, Cf32, Cf16, ldc,
      bias, resid, M, N, K, alpha, epi, causal, vt, binner, aO, aI, bO, bI, cO, cI);
}

extern "C" void kernel_launch(void* const* d_in, const int* in_sizes, int n_in,
                              void* d_out, int out_size, void* d_ws, size_t ws_size,
                              hipStream_t stream)
{
  (void)in_sizes; (void)n_in; (void)out_size; (void)ws_size;
  const float* word_emb = (const float*)d_in[0];
  const float* pos_emb  = (const float*)d_in[1];
  const float* type_emb = (const float*)d_in[2];
  const float* emb_ln_g = (const float*)d_in[3];
  const float* emb_ln_b = (const float*)d_in[4];
  const float* Wq = (const float*)d_in[5];   const float* bq = (const float*)d_in[6];
  const float* Wk = (const float*)d_in[7];   const float* bk = (const float*)d_in[8];
  const float* Wv = (const float*)d_in[9];   const float* bv = (const float*)d_in[10];
  const float* Wo = (const float*)d_in[11];  const float* bo = (const float*)d_in[12];
  const float* ln1_g = (const float*)d_in[13]; const float* ln1_b = (const float*)d_in[14];
  const float* W1 = (const float*)d_in[15];  const float* b1 = (const float*)d_in[16];
  const float* W2 = (const float*)d_in[17];  const float* b2 = (const float*)d_in[18];
  const float* ln2_g = (const float*)d_in[19]; const float* ln2_b = (const float*)d_in[20];
  const float* cls_W = (const float*)d_in[21]; const float* cls_b = (const float*)d_in[22];
  const int* xids = (const int*)d_in[23];
  const int* yids = (const int*)d_in[24];

  char* ws = (char*)d_ws;
  size_t off = 0;
  auto alloc = [&](size_t bytes) -> char* {
    char* p = ws + off; off += (bytes + 255) & ~(size_t)255; return p;
  };
  half_t* wqT   = (half_t*)alloc((size_t)LL * HH * HH * 2);
  half_t* wkT   = (half_t*)alloc((size_t)LL * HH * HH * 2);
  half_t* wvT   = (half_t*)alloc((size_t)LL * HH * HH * 2);
  half_t* woT   = (half_t*)alloc((size_t)LL * HH * HH * 2);
  half_t* w1T   = (half_t*)alloc((size_t)LL * HH * FFD * 2);
  half_t* w2T   = (half_t*)alloc((size_t)LL * FFD * HH * 2);
  half_t* clsWT = (half_t*)alloc((size_t)VOCAB * HH * 2);
  float*  h32   = (float* )alloc((size_t)MM * HH * 4);
  half_t* h16   = (half_t*)alloc((size_t)MM * HH * 2);
  float*  h2_32 = (float* )alloc((size_t)MM * HH * 4);
  half_t* h2_16 = (half_t*)alloc((size_t)MM * HH * 2);
  float*  res32 = (float* )alloc((size_t)MM * HH * 4);
  half_t* q16   = (half_t*)alloc((size_t)MM * HH * 2);
  half_t* k16   = (half_t*)alloc((size_t)MM * HH * 2);
  half_t* vT16  = (half_t*)alloc((size_t)MM * HH * 2);
  half_t* ctx16 = (half_t*)alloc((size_t)MM * HH * 2);
  float*  sc32  = (float* )alloc((size_t)BB * NHEAD * SS * SS * 4);
  half_t* at16  = (half_t*)alloc((size_t)BB * NHEAD * SS * SS * 2);
  half_t* ff16  = (half_t*)alloc((size_t)MM * FFD * 2);
  float*  nllb  = (float* )alloc((size_t)MM * 4);

  // 1) one-time weight convert+transpose to f16 [N][K]
  dim3 tb(32, 8);
  long long sHH = (long long)HH * HH, sHF = (long long)HH * FFD;
  transpose_f32_to_f16<<<dim3(HH/32, HH/32, LL), tb, 0, stream>>>(Wq, wqT, HH, HH, sHH, sHH);
  transpose_f32_to_f16<<<dim3(HH/32, HH/32, LL), tb, 0, stream>>>(Wk, wkT, HH, HH, sHH, sHH);
  transpose_f32_to_f16<<<dim3(HH/32, HH/32, LL), tb, 0, stream>>>(Wv, wvT, HH, HH, sHH, sHH);
  transpose_f32_to_f16<<<dim3(HH/32, HH/32, LL), tb, 0, stream>>>(Wo, woT, HH, HH, sHH, sHH);
  transpose_f32_to_f16<<<dim3(FFD/32, HH/32, LL), tb, 0, stream>>>(W1, w1T, HH, FFD, sHF, sHF);
  transpose_f32_to_f16<<<dim3(HH/32, FFD/32, LL), tb, 0, stream>>>(W2, w2T, FFD, HH, sHF, sHF);
  transpose_f32_to_f16<<<dim3((VOCAB+31)/32, HH/32, 1), tb, 0, stream>>>(cls_W, clsWT, HH, VOCAB, 0, 0);

  // 2) embeddings + LN
  embed_kernel<<<dim3(MM), dim3(256), 0, stream>>>(xids, word_emb, pos_emb, type_emb, res32);
  layernorm_kernel<<<dim3(MM/8), dim3(256), 0, stream>>>(res32, emb_ln_g, emb_ln_b, h32, h16, MM, HH);

  // 3) transformer layers
  for (int l = 0; l < LL; ++l) {
    const half_t* wq_l = wqT + (size_t)l * HH * HH;
    const half_t* wk_l = wkT + (size_t)l * HH * HH;
    const half_t* wv_l = wvT + (size_t)l * HH * HH;
    const half_t* wo_l = woT + (size_t)l * HH * HH;
    const half_t* w1_l = w1T + (size_t)l * HH * FFD;
    const half_t* w2_l = w2T + (size_t)l * FFD * HH;

    // Q, K projections
    launch_gemm(stream, h16, HH, wq_l, HH, nullptr, q16, HH, bq + (size_t)l*HH, nullptr,
                MM, HH, HH, 1.f, 0, 0, 0, 1, 1, 0,0,0,0,0,0);
    launch_gemm(stream, h16, HH, wk_l, HH, nullptr, k16, HH, bk + (size_t)l*HH, nullptr,
                MM, HH, HH, 1.f, 0, 0, 0, 1, 1, 0,0,0,0,0,0);
    // V projection written transposed as V^T[b][h][d][s]
    launch_gemm(stream, h16, HH, wv_l, HH, nullptr, vT16, HH, bv + (size_t)l*HH, nullptr,
                MM, HH, HH, 1.f, 0, 0, 1, 1, 1, 0,0,0,0,0,0);
    // scores = QK^T/8 + causal mask   (batched over b*h = 96)
    launch_gemm(stream, q16, HH, k16, HH, sc32, nullptr, SS, nullptr, nullptr,
                SS, SS, DHEAD, 0.125f, 0, 1, 0, BB*NHEAD, NHEAD,
                (long long)SS*HH, DHEAD, (long long)SS*HH, DHEAD,
                (long long)NHEAD*SS*SS, (long long)SS*SS);
    softmax_kernel<<<dim3(BB*NHEAD*SS/8), dim3(256), 0, stream>>>(sc32, at16, BB*NHEAD*SS, SS);
    // ctx = attn @ V  (B^T = V^T), scattered back to [token][h*64+d]
    launch_gemm(stream, at16, SS, vT16, SS, nullptr, ctx16, HH, nullptr, nullptr,
                SS, DHEAD, SS, 1.f, 0, 0, 0, BB*NHEAD, NHEAD,
                (long long)NHEAD*SS*SS, (long long)SS*SS,
                (long long)NHEAD*DHEAD*SS, (long long)DHEAD*SS,
                (long long)SS*HH, DHEAD);
    // out-proj + bias + residual(h)
    launch_gemm(stream, ctx16, HH, wo_l, HH, res32, nullptr, HH, bo + (size_t)l*HH, h32,
                MM, HH, HH, 1.f, 2, 0, 0, 1, 1, 0,0,0,0,0,0);
    layernorm_kernel<<<dim3(MM/8), dim3(256), 0, stream>>>(res32, ln1_g + (size_t)l*HH, ln1_b + (size_t)l*HH, h2_32, h2_16, MM, HH);
    // FFN
    launch_gemm(stream, h2_16, HH, w1_l, HH, nullptr, ff16, FFD, b1 + (size_t)l*FFD, nullptr,
                MM, FFD, HH, 1.f, 1, 0, 0, 1, 1, 0,0,0,0,0,0);
    launch_gemm(stream, ff16, FFD, w2_l, FFD, res32, nullptr, HH, b2 + (size_t)l*HH, h2_32,
                MM, HH, FFD, 1.f, 2, 0, 0, 1, 1, 0,0,0,0,0,0);
    layernorm_kernel<<<dim3(MM/8), dim3(256), 0, stream>>>(res32, ln2_g + (size_t)l*HH, ln2_b + (size_t)l*HH, h32, h16, MM, HH);
  }

  // 4) fused classifier + online log-sum-exp NLL, then deterministic mean
  cls_nll_kernel<<<dim3(MM/16), dim3(256), 0, stream>>>(h16, clsWT, cls_b, yids, nllb, HH, VOCAB);
  reduce_mean_kernel<<<dim3(1), dim3(256), 0, stream>>>(nllb, (float*)d_out, MM);
}
